// GAT_24464133718500
// MI455X (gfx1250) — compile-verified, matrix operand order checked by reading, not run on hardware
//
#include <hip/hip_runtime.h>
#include <hip/hip_bf16.h>

typedef _Float16 half_t;
typedef _Float16 v16h __attribute__((ext_vector_type(16)));
typedef float    v8f  __attribute__((ext_vector_type(8)));

// ---------------------------------------------------------------------------
// Pack int32 adjacency into bitmasks: bits[row][w] bit b = adj[row][w*32+b]!=0
// ---------------------------------------------------------------------------
__global__ void k_pack_adj(const int* __restrict__ adj, unsigned* __restrict__ bits,
                           int N, int Wd) {
    int gid  = blockIdx.x * blockDim.x + threadIdx.x;
    int wid  = gid >> 5;
    int lane = gid & 31;
    if (wid >= N * Wd) return;
    int row = wid / Wd, w = wid - row * Wd;
    int j = (w << 5) + lane;
    int v = adj[(long long)row * N + j];
    unsigned long long b = __ballot(v != 0);
    if (lane == 0) bits[(long long)row * Wd + w] = (unsigned)b;
}

// ---------------------------------------------------------------------------
// LayerNorm: one block per row, blockDim == F (multiple of 32, <= 256)
// ---------------------------------------------------------------------------
__global__ void k_layernorm(const float* __restrict__ x, const float* __restrict__ g,
                            const float* __restrict__ b, float* __restrict__ y, int F) {
    int n = blockIdx.x, t = threadIdx.x;
    float v = x[(long long)n * F + t];
    float s = v, s2 = v * v;
    for (int off = 16; off; off >>= 1) {
        s  += __shfl_xor(s,  off, 32);
        s2 += __shfl_xor(s2, off, 32);
    }
    __shared__ float sh[8], sh2[8];
    int wid = t >> 5, ln = t & 31;
    if (ln == 0) { sh[wid] = s; sh2[wid] = s2; }
    __syncthreads();
    int nw = F >> 5;
    if (t < 32) {
        s  = (ln < nw) ? sh[ln]  : 0.f;
        s2 = (ln < nw) ? sh2[ln] : 0.f;
        for (int off = 4; off; off >>= 1) {
            s  += __shfl_xor(s,  off, 32);
            s2 += __shfl_xor(s2, off, 32);
        }
        if (ln == 0) { sh[0] = s; sh2[0] = s2; }
    }
    __syncthreads();
    float mean = sh[0] / (float)F;
    float var  = sh2[0] / (float)F - mean * mean;
    y[(long long)n * F + t] = (v - mean) * rsqrtf(var + 1e-5f) * g[t] + b[t];
}

// ---------------------------------------------------------------------------
// Per-node linear: h[n, H*Dh] = x[n,:] @ W  (W layout [Fin][H][Dh]).
// Writes f16 h directly PRE-SWIZZLED into the WMMA B-fragment lane layout:
//   bswz[(((head*NT + tile)*nchunk + K-chunk)*32 + lane)*16 + t]
// with lane = col + 16*((K>>3)&1), t = (K&7) + 8*(K>=16), col = d&15,
// so each lane's 16 B halfs for a chunk are one contiguous 32B load.
// Cols >= Dh within a 16-wide tile are zero (masked padding).
// Also emits s_src[h][n] = h.a_src, s_dst[h][n] = h.a_dst.
// blockDim = 64, one block per node.
// ---------------------------------------------------------------------------
__global__ void k_linear(const float* __restrict__ x, const float* __restrict__ W,
                         const float* __restrict__ asrc, const float* __restrict__ adst,
                         half_t* __restrict__ bswz, float* __restrict__ ssrc,
                         float* __restrict__ sdst,
                         int N, int Fin, int H, int Dh, int DhPad) {
    int n = blockIdx.x, od = threadIdx.x;
    int HD = H * Dh;
    int NT = DhPad >> 4;
    int nchunk = N >> 5;
    __shared__ float sh[64];
    if (od < HD) {
        const float* xr = x + (long long)n * Fin;
        float acc = 0.f;
        for (int f = 0; f < Fin; ++f) acc += xr[f] * W[f * HD + od];
        sh[od] = acc;
    }
    __syncthreads();

    int K  = n & 31;           // position of this node inside its K-chunk
    int c  = n >> 5;           // K-chunk index
    int hi = (K >> 3) & 1;
    int tt = (K & 7) + ((K >> 4) << 3);
    for (int idx = od; idx < H * DhPad; idx += 64) {
        int hh   = idx / DhPad, d = idx - hh * DhPad;
        int tile = d >> 4, col = d & 15;
        int lane = col + (hi << 4);
        float v  = (d < Dh) ? sh[hh * Dh + d] : 0.f;
        size_t base = ((((size_t)(hh * NT + tile) * nchunk + c) << 5) + lane);
        bswz[base * 16 + tt] = (half_t)v;
    }
    if (od < H) {
        float s1 = 0.f, s2 = 0.f;
        for (int d = 0; d < Dh; ++d) {
            float hv = sh[od * Dh + d];
            s1 += hv * asrc[od * Dh + d];
            s2 += hv * adst[od * Dh + d];
        }
        ssrc[(long long)od * N + n] = s1;
        sdst[(long long)od * N + n] = s2;
    }
}

// ---------------------------------------------------------------------------
// Attention pass 1: per (head,row) wave computes masked row max and 1/Z.
// Bits word per iteration is wave-uniform -> scalar load; all math is
// branchless (cndmask select / mask multiply), no exec-mask branching.
// ---------------------------------------------------------------------------
__global__ void k_attn_pass1(const unsigned* __restrict__ bits, const float* __restrict__ ssrc,
                             const float* __restrict__ sdst, float* __restrict__ rowmax,
                             float* __restrict__ invZ, int N, int Wd, int H, float alpha) {
    int gid = blockIdx.x * blockDim.x + threadIdx.x;
    int wid = gid >> 5, lane = gid & 31;
    if (wid >= H * N) return;
    int head = wid / N, row = wid - head * N;
    const unsigned* br = bits + (long long)row * Wd;
    const float* sd = sdst + (long long)head * N;
    float si = ssrc[(long long)head * N + row];

    float m = -3.0e38f;
    for (int it = 0; it < Wd; ++it) {
        unsigned w = br[it];                       // uniform across wave
        float e = si + sd[(it << 5) + lane];
        e = (e > 0.f) ? e : alpha * e;
        float ee = ((w >> lane) & 1u) ? e : -3.0e38f;   // cndmask, branchless
        m = fmaxf(m, ee);
    }
    for (int off = 16; off; off >>= 1) m = fmaxf(m, __shfl_xor(m, off, 32));

    float z = 0.f;
    for (int it = 0; it < Wd; ++it) {
        unsigned w = br[it];
        float e = si + sd[(it << 5) + lane];
        e = (e > 0.f) ? e : alpha * e;
        float msk = (float)((w >> lane) & 1u);
        z += msk * __expf(fminf(e - m, 0.f));      // mask-multiply, branchless
    }
    for (int off = 16; off; off >>= 1) z += __shfl_xor(z, off, 32);

    if (lane == 0) {
        rowmax[wid] = m;
        invZ[wid]   = (z > 0.f) ? (1.f / z) : 0.f;
    }
}

// ---------------------------------------------------------------------------
// Attention pass 2 (WMMA core). One wave per (head, 16-row tile).
// P(16x32) f16 generated on the fly (packed adj + rank-1 scores, 2-pass
// softmax normalization), B fragments are single 32B vector loads from the
// pre-swizzled buffer, f32 16x16 accumulation via v_wmma_f32_16x16x32_f16.
// NTILE/MODE compile-time: fixed-VGPR accumulators, fully unrolled tiles.
// Probability generation is fully branchless: mask becomes a multiply,
// exponent argument clamped to <= 0 so masked lanes can't overflow.
// s_dst chunk values fetched via lane shuffles (ds_bpermute).
// MODE 0: fused ELU + head-concat write of x_next (and optional seq slot)
// MODE 1: fused ELU + sum over nodes via f32 atomics into ysum
// ---------------------------------------------------------------------------
template <int NTILE, int MODE>
__global__ void k_attn_apply(const unsigned* __restrict__ bits, const float* __restrict__ ssrc,
                             const float* __restrict__ sdst, const float* __restrict__ rowmax,
                             const float* __restrict__ invZ, const half_t* __restrict__ bswz,
                             float* __restrict__ xout, float* __restrict__ seqout,
                             float* __restrict__ ysum,
                             int N, int Wd, int Dh, float alpha, int HD) {
    int nblk = N >> 4;
    int nchunk = N >> 5;
    int head = blockIdx.x / nblk;
    int blk  = blockIdx.x - head * nblk;
    int lane = threadIdx.x;
    int r    = lane & 15;
    int i    = (blk << 4) + r;
    int koff = (lane & 16) >> 1;            // 0 for lanes 0-15, 8 for 16-31

    float si = ssrc[(long long)head * N + i];
    float mi = rowmax[(long long)head * N + i];
    float zi = invZ[(long long)head * N + i];
    const float*    sd = sdst + (long long)head * N;
    const unsigned* br = bits + (long long)i * Wd;

    const v8f vzero = {0.f, 0.f, 0.f, 0.f, 0.f, 0.f, 0.f, 0.f};
    v8f acc[NTILE];
#pragma unroll
    for (int t = 0; t < NTILE; ++t) acc[t] = vzero;

    for (int c = 0; c < nchunk; ++c) {
        int k0 = c << 5;
        unsigned w = br[c];
        float sdv = sd[k0 + lane];          // one coalesced load; rest via permute
        if (c + 1 < nchunk) __builtin_prefetch(&br[c + 1], 0, 1);

        v16h a;
#pragma unroll
        for (int t = 0; t < 16; ++t) {
            int K = t + (t & 8) + koff;
            float e = si + __shfl(sdv, K, 32);
            e = (e > 0.f) ? e : alpha * e;                 // LeakyReLU (cndmask)
            float msk = (float)((w >> K) & 1u);
            float p = msk * __expf(fminf(e - mi, 0.f)) * zi;   // branchless
            a[t] = (half_t)p;
        }
#pragma unroll
        for (int tile = 0; tile < NTILE; ++tile) {
            size_t base = ((((size_t)(head * NTILE + tile) * nchunk + c) << 5) + lane) * 16;
            v16h b = *(const v16h*)(bswz + base);   // 32B aligned vector load
            acc[tile] = __builtin_amdgcn_wmma_f32_16x16x32_f16(
                false, a, false, b, (short)0, acc[tile], false, false);
        }
    }

    if (MODE == 0) {
        int col = lane & 15;
        if (col < Dh) {
#pragma unroll
            for (int v = 0; v < 8; ++v) {
                int M = v + koff;
                int n = (blk << 4) + M;
                float val = acc[0][v];
                val = (val > 0.f) ? val : (__expf(fminf(val, 0.f)) - 1.f);   // ELU
                int feat = head * Dh + col;
                xout[(long long)n * HD + feat] = val;
                if (seqout) seqout[(long long)n * HD + feat] = val;
            }
        }
    } else {
#pragma unroll
        for (int tile = 0; tile < NTILE; ++tile) {
            int col = (lane & 15) + (tile << 4);
            float s = 0.f;
#pragma unroll
            for (int v = 0; v < 8; ++v) {
                float val = acc[tile][v];
                val = (val > 0.f) ? val : (__expf(fminf(val, 0.f)) - 1.f);   // ELU
                s += val;
            }
            s += __shfl_xor(s, 16, 32);                          // rows 0-7 + 8-15
            if (lane < 16) atomicAdd(&ysum[col], s);
        }
    }
}

// ---------------------------------------------------------------------------
// Per-node LSTM (torch gate order i,f,g,o). seq layout [T][N][H]. H <= 12.
// ---------------------------------------------------------------------------
__device__ __forceinline__ float sigmf(float x) { return 1.f / (1.f + __expf(-x)); }

__global__ void k_lstm(const float* __restrict__ seq, const float* __restrict__ Wih,
                       const float* __restrict__ Whh, const float* __restrict__ bih,
                       const float* __restrict__ bhh, float* __restrict__ hout,
                       int N, int Hh, int T) {
    __shared__ float sWih[576], sWhh[576], sbih[48], sbhh[48];
    int W4 = 4 * Hh * Hh;
    for (int idx = threadIdx.x; idx < W4; idx += blockDim.x) {
        sWih[idx] = Wih[idx];
        sWhh[idx] = Whh[idx];
    }
    for (int idx = threadIdx.x; idx < 4 * Hh; idx += blockDim.x) {
        sbih[idx] = bih[idx];
        sbhh[idx] = bhh[idx];
    }
    __syncthreads();
    int n = blockIdx.x * blockDim.x + threadIdx.x;
    if (n >= N) return;

    float h[12], c[12], xr[12], gates[48];
    for (int d = 0; d < Hh; ++d) { h[d] = 0.f; c[d] = 0.f; }
    for (int t = 0; t < T; ++t) {
        const float* xt = seq + ((long long)t * N + n) * Hh;
        for (int j = 0; j < Hh; ++j) xr[j] = xt[j];
        for (int k = 0; k < 4 * Hh; ++k) {
            float g = sbih[k] + sbhh[k];
            for (int j = 0; j < Hh; ++j)
                g += xr[j] * sWih[k * Hh + j] + h[j] * sWhh[k * Hh + j];
            gates[k] = g;
        }
        for (int d = 0; d < Hh; ++d) {
            float iv = sigmf(gates[d]);
            float fv = sigmf(gates[Hh + d]);
            float gv = tanhf(gates[2 * Hh + d]);
            float ov = sigmf(gates[3 * Hh + d]);
            c[d] = fv * c[d] + iv * gv;
            h[d] = ov * tanhf(c[d]);
        }
    }
    for (int d = 0; d < Hh; ++d) hout[(long long)n * Hh + d] = h[d];
}

__global__ void k_zero(float* __restrict__ p, int n) {
    int i = blockIdx.x * blockDim.x + threadIdx.x;
    if (i < n) p[i] = 0.f;
}

// out[o] = relu(fcb[o] + sum_k y[k] * fcW[o,k]),  fcW: [64][128]
__global__ void k_fc(const float* __restrict__ y, const float* __restrict__ W,
                     const float* __restrict__ b, float* __restrict__ out) {
    int o = threadIdx.x;   // 64 threads
    float s = b[o];
    for (int k = 0; k < 128; ++k) s += y[k] * W[o * 128 + k];
    out[o] = fmaxf(s, 0.f);
}

// ---------------------------------------------------------------------------
// Host orchestration
// ---------------------------------------------------------------------------
extern "C" void kernel_launch(void* const* d_in, const int* in_sizes, int n_in,
                              void* d_out, int out_size, void* d_ws, size_t ws_size,
                              hipStream_t stream) {
    (void)in_sizes; (void)n_in; (void)out_size; (void)ws_size;

    // setup_inputs order, params flattened alphabetically (jax pytree order)
    const float* nf1   = (const float*)d_in[0];
    const int*   adj1  = (const int*)  d_in[1];
    const float* nf2   = (const float*)d_in[2];
    const int*   adj2  = (const int*)  d_in[3];
    const float* fcW   = (const float*)d_in[4];
    const float* fcb   = (const float*)d_in[5];
    const float* j1W   = (const float*)d_in[6];
    const float* j1ad  = (const float*)d_in[7];
    const float* j1as  = (const float*)d_in[8];
    const float* j2W   = (const float*)d_in[9];
    const float* j2ad  = (const float*)d_in[10];
    const float* j2as  = (const float*)d_in[11];
    const float* ln1b  = (const float*)d_in[12];
    const float* ln1g  = (const float*)d_in[13];
    const float* ln2b  = (const float*)d_in[14];
    const float* ln2g  = (const float*)d_in[15];
    const float* l1Whh = (const float*)d_in[16];
    const float* l1Wih = (const float*)d_in[17];
    const float* l1bhh = (const float*)d_in[18];
    const float* l1bih = (const float*)d_in[19];
    const float* l2Whh = (const float*)d_in[20];
    const float* l2Wih = (const float*)d_in[21];
    const float* l2bhh = (const float*)d_in[22];
    const float* l2bih = (const float*)d_in[23];
    const float* o1W   = (const float*)d_in[24];
    const float* o1ad  = (const float*)d_in[25];
    const float* o1as  = (const float*)d_in[26];
    const float* o2W   = (const float*)d_in[27];
    const float* o2ad  = (const float*)d_in[28];
    const float* o2as  = (const float*)d_in[29];
    const float* p1W   = (const float*)d_in[30];
    const float* p1ad  = (const float*)d_in[31];
    const float* p1as  = (const float*)d_in[32];
    const float* p2W   = (const float*)d_in[33];
    const float* p2ad  = (const float*)d_in[34];
    const float* p2as  = (const float*)d_in[35];

    const int N1 = 3072, F1 = 128, N2 = 2048, F2 = 64;
    const int Wd1 = N1 / 32, Wd2 = N2 / 32;

    char* wp = (char*)d_ws;
    auto carve = [&](size_t bytes) -> void* {
        void* r = (void*)wp;
        wp += (bytes + 255) & ~(size_t)255;
        return r;
    };
    unsigned* bits1 = (unsigned*)carve((size_t)N1 * Wd1 * 4);
    unsigned* bits2 = (unsigned*)carve((size_t)N2 * Wd2 * 4);
    float* xln1  = (float*)carve((size_t)N1 * F1 * 4);
    float* xln2  = (float*)carve((size_t)N2 * F2 * 4);
    float* x1    = (float*)carve((size_t)N1 * 12 * 4);
    float* x2    = (float*)carve((size_t)N2 * 6 * 4);
    float* seq1  = (float*)carve((size_t)5 * N1 * 12 * 4);
    float* seq2  = (float*)carve((size_t)5 * N2 * 6 * 4);
    half_t* bswz1 = (half_t*)carve((size_t)N1 * 64 * 2);   // covers 3 heads*16 and 1*64
    half_t* bswz2 = (half_t*)carve((size_t)N2 * 64 * 2);
    float* ss1 = (float*)carve((size_t)3 * N1 * 4);
    float* sd1 = (float*)carve((size_t)3 * N1 * 4);
    float* rm1 = (float*)carve((size_t)3 * N1 * 4);
    float* iz1 = (float*)carve((size_t)3 * N1 * 4);
    float* ss2 = (float*)carve((size_t)3 * N2 * 4);
    float* sd2 = (float*)carve((size_t)3 * N2 * 4);
    float* rm2 = (float*)carve((size_t)3 * N2 * 4);
    float* iz2 = (float*)carve((size_t)3 * N2 * 4);
    float* lh1 = (float*)carve((size_t)N1 * 12 * 4);
    float* lh2 = (float*)carve((size_t)N2 * 6 * 4);
    float* yv  = (float*)carve(128 * 4);

    auto gat = [&](const unsigned* bits, const float* xin,
                   const float* W, const float* as, const float* ad,
                   int N, int Wd, int Fin, int H, int Dh, int DhPad, float alpha,
                   half_t* bswz, float* ssrc, float* sdst, float* rmax, float* izn,
                   float* xout, float* seqout, float* ysum, int mode) {
        k_linear<<<N, 64, 0, stream>>>(xin, W, as, ad, bswz, ssrc, sdst, N, Fin, H, Dh, DhPad);
        int waves = H * N;
        k_attn_pass1<<<(waves * 32 + 127) / 128, 128, 0, stream>>>(
            bits, ssrc, sdst, rmax, izn, N, Wd, H, alpha);
        if (mode == 0)
            k_attn_apply<1, 0><<<H * (N / 16), 32, 0, stream>>>(
                bits, ssrc, sdst, rmax, izn, bswz, xout, seqout, ysum, N, Wd, Dh, alpha, H * Dh);
        else
            k_attn_apply<4, 1><<<H * (N / 16), 32, 0, stream>>>(
                bits, ssrc, sdst, rmax, izn, bswz, xout, seqout, ysum, N, Wd, Dh, alpha, H * Dh);
    };

    // ---- prologue ----
    k_pack_adj<<<(N1 * Wd1 * 32 + 255) / 256, 256, 0, stream>>>(adj1, bits1, N1, Wd1);
    k_pack_adj<<<(N2 * Wd2 * 32 + 255) / 256, 256, 0, stream>>>(adj2, bits2, N2, Wd2);
    k_layernorm<<<N1, F1, 0, stream>>>(nf1, ln1g, ln1b, xln1, F1);
    k_layernorm<<<N2, F2, 0, stream>>>(nf2, ln2g, ln2b, xln2, F2);
    k_zero<<<1, 128, 0, stream>>>(yv, 128);

    // ---- graph 1 ----
    gat(bits1, xln1, p1W, p1as, p1ad, N1, Wd1, F1, 1, 12, 16, 0.01f,
        bswz1, ss1, sd1, rm1, iz1, x1, nullptr, nullptr, 0);
    for (int t = 0; t < 5; ++t)
        gat(bits1, x1, j1W, j1as, j1ad, N1, Wd1, 12, 3, 4, 16, 0.01f,
            bswz1, ss1, sd1, rm1, iz1, x1, seq1 + (size_t)t * N1 * 12, nullptr, 0);
    k_lstm<<<(N1 + 255) / 256, 256, 0, stream>>>(seq1, l1Wih, l1Whh, l1bih, l1bhh, lh1, N1, 12, 5);
    gat(bits1, lh1, o1W, o1as, o1ad, N1, Wd1, 12, 1, 64, 64, 0.001f,
        bswz1, ss1, sd1, rm1, iz1, nullptr, nullptr, yv + 0, 1);

    // ---- graph 2 ----
    gat(bits2, xln2, p2W, p2as, p2ad, N2, Wd2, F2, 1, 6, 16, 0.01f,
        bswz2, ss2, sd2, rm2, iz2, x2, nullptr, nullptr, 0);
    for (int t = 0; t < 5; ++t)
        gat(bits2, x2, j2W, j2as, j2ad, N2, Wd2, 6, 3, 2, 16, 0.01f,
            bswz2, ss2, sd2, rm2, iz2, x2, seq2 + (size_t)t * N2 * 6, nullptr, 0);
    k_lstm<<<(N2 + 255) / 256, 256, 0, stream>>>(seq2, l2Wih, l2Whh, l2bih, l2bhh, lh2, N2, 6, 5);
    gat(bits2, lh2, o2W, o2as, o2ad, N2, Wd2, 6, 1, 64, 64, 0.001f,
        bswz2, ss2, sd2, rm2, iz2, nullptr, nullptr, yv + 64, 1);

    // ---- final FC ----
    k_fc<<<1, 64, 0, stream>>>(yv, fcW, fcb, (float*)d_out);
}
